// Attention_66898410602517
// MI455X (gfx1250) — compile-verified, matrix operand order checked by reading, not run on hardware
//
#include <hip/hip_runtime.h>

typedef __attribute__((ext_vector_type(16))) _Float16 v16h;
typedef __attribute__((ext_vector_type(8)))  _Float16 v8h;
typedef __attribute__((ext_vector_type(8)))  float    v8f;
typedef __attribute__((ext_vector_type(4)))  float    v4f;

#define BATCH 16
#define SEQ   4096
#define DM    128          // dmodel == head size

#define SM_SCALE 0.08838834764831845f   // 1/sqrt(128)
#define LOG2E    1.4426950408889634f

#define BM   128           // query rows per block (8 waves x 16)
#define BN   64            // keys per tile (amortizes softmax VALU)
#define PADH 8             // LDS padding (halves) to break bank conflicts

// Load a 16x32 f16 fragment (A-layout / symmetric B-layout) from a row-major
// f16 matrix: lane owns row lane%16; halves 0..7 at k=d0+hi*8, halves
// 8..15 at k=d0+16+hi*8. Two 16-byte loads per lane.
__device__ __forceinline__ v16h frag_ld(const _Float16* row_base, int stride_h,
                                        int d0, int lane)
{
    int r  = lane & 15;
    int hi = lane >> 4;
    const _Float16* p = row_base + (size_t)r * stride_h + d0 + hi * 8;
    v8h lo = *(const v8h*)(p);
    v8h hv = *(const v8h*)(p + 16);
    return __builtin_shufflevector(lo, hv, 0,1,2,3,4,5,6,7,8,9,10,11,12,13,14,15);
}

__device__ __forceinline__ v8f wmma_f16(v16h a, v16h b, v8f c)
{
    return __builtin_amdgcn_wmma_f32_16x16x32_f16(
        /*neg_a=*/false, a, /*neg_b=*/false, b,
        /*c_mod=*/(short)0, c, /*reuse_a=*/false, /*reuse_b=*/false);
}

// Reductions across the 16-lane half that holds one matrix row (xor<16 stays
// within each half of the wave32).
__device__ __forceinline__ float rmax16(float v)
{
    v = fmaxf(v, __shfl_xor(v, 1));
    v = fmaxf(v, __shfl_xor(v, 2));
    v = fmaxf(v, __shfl_xor(v, 4));
    v = fmaxf(v, __shfl_xor(v, 8));
    return v;
}
__device__ __forceinline__ float rsum16(float v)
{
    v += __shfl_xor(v, 1);
    v += __shfl_xor(v, 2);
    v += __shfl_xor(v, 4);
    v += __shfl_xor(v, 8);
    return v;
}

// ---------------------------------------------------------------------------
// Kernel 1: QKV projection, fp32 x @ fp32 W -> f16 Q/K/V via WMMA
// block = 256 threads (8 waves), 128 rows of x per block
// ---------------------------------------------------------------------------
__global__ __launch_bounds__(256)
void qkv_proj_kernel(const float* __restrict__ x,
                     const float* __restrict__ Wq,
                     const float* __restrict__ Wk,
                     const float* __restrict__ Wv,
                     _Float16* __restrict__ Q,
                     _Float16* __restrict__ K,
                     _Float16* __restrict__ V)
{
    __shared__ _Float16 sWt[DM][DM + PADH];   // W^T: [h][d], padded stride

    const int tid  = threadIdx.x;
    const int lane = tid & 31;
    const int wv   = tid >> 5;                 // wave 0..7
    const size_t row0 = (size_t)blockIdx.x * 128;
    const int r0w = wv * 16;

    // Load this wave's 16 x-rows as 4 A-fragments (f32 -> f16)
    v16h xf[4];
    {
        int r = lane & 15, hi = lane >> 4;
        const float* xp = x + (row0 + r0w + r) * (size_t)DM;
#pragma unroll
        for (int ks = 0; ks < 4; ++ks) {
            int d0 = ks * 32 + hi * 8;
            v4f a0 = *(const v4f*)(xp + d0);
            v4f a1 = *(const v4f*)(xp + d0 + 4);
            v4f b0 = *(const v4f*)(xp + d0 + 16);
            v4f b1 = *(const v4f*)(xp + d0 + 20);
            v16h f;
#pragma unroll
            for (int i = 0; i < 4; ++i) {
                f[i]      = (_Float16)a0[i];
                f[i + 4]  = (_Float16)a1[i];
                f[i + 8]  = (_Float16)b0[i];
                f[i + 12] = (_Float16)b1[i];
            }
            xf[ks] = f;
        }
    }

    const float* Ws[3] = { Wq, Wk, Wv };
    _Float16*    Os[3] = { Q, K, V };

#pragma unroll
    for (int m = 0; m < 3; ++m) {
        __syncthreads();
        const float* W = Ws[m];
        // stage W^T into LDS as f16
        for (int e = tid; e < DM * DM; e += 256) {
            int d = e >> 7, h = e & 127;
            sWt[h][d] = (_Float16)W[e];
        }
        __syncthreads();

        _Float16* Om = Os[m];
        const int hi = lane >> 4;
#pragma unroll
        for (int sub = 0; sub < 8; ++sub) {
            v8f acc = {};
#pragma unroll
            for (int ks = 0; ks < 4; ++ks) {
                v16h bf = frag_ld(&sWt[sub * 16][0], DM + PADH, ks * 32, lane);
                acc = wmma_f16(xf[ks], bf, acc);
            }
            int c = sub * 16 + (lane & 15);
#pragma unroll
            for (int v = 0; v < 8; ++v) {
                size_t gr = row0 + r0w + hi * 8 + v;
                Om[gr * DM + c] = (_Float16)acc[v];
            }
        }
    }
}

// ---------------------------------------------------------------------------
// Kernel 2: causal flash attention, f16 WMMA, f32 accumulation
// block = 256 threads (8 waves) = 128 query rows; key tiles of 64
// ---------------------------------------------------------------------------
__global__ __launch_bounds__(256)
void attn_kernel(const _Float16* __restrict__ Q,
                 const _Float16* __restrict__ K,
                 const _Float16* __restrict__ V,
                 float* __restrict__ out)
{
    __shared__ _Float16 sK [BN][DM + PADH];        // K tile, row-major [key][d]
    __shared__ _Float16 sVt[DM][BN + PADH];        // V tile transposed [d][key]
    __shared__ _Float16 sP [8][16][BN + PADH];     // per-wave P staging

    const int tid  = threadIdx.x;
    const int lane = tid & 31;
    const int wv   = tid >> 5;
    const int b    = blockIdx.y;
    const int q0   = blockIdx.x * BM;
    const int i0w  = q0 + wv * 16;
    const int r    = lane & 15;
    const int hi   = lane >> 4;

    const _Float16* Qb = Q + ((size_t)b * SEQ + i0w) * DM;
    const _Float16* Kb = K + (size_t)b * SEQ * DM;
    const _Float16* Vb = V + (size_t)b * SEQ * DM;

    // Resident Q fragments: 16 rows x 128 head-dim
    v16h qf[4];
#pragma unroll
    for (int ks = 0; ks < 4; ++ks) qf[ks] = frag_ld(Qb, DM, ks * 32, lane);

    v8f o[8];
    v8f vzero = {};
#pragma unroll
    for (int s = 0; s < 8; ++s) o[s] = vzero;
    float mrow[8], lrow[8];
#pragma unroll
    for (int v = 0; v < 8; ++v) { mrow[v] = -__builtin_inff(); lrow[v] = 0.0f; }

    const float sscale = SM_SCALE * LOG2E;   // exp2-domain softmax
    const int kend = q0 + BM;

    for (int j0 = 0; j0 < kend; j0 += BN) {
        // --- cooperative stage: K row-major, V transposed ---
        {
            int krow = tid >> 2;             // 0..63
            int dc   = (tid & 3) * 32;       // 0,32,64,96
            const _Float16* kp = Kb + ((size_t)(j0 + krow)) * DM + dc;
            v8h k0 = *(const v8h*)kp;
            v8h k1 = *(const v8h*)(kp + 8);
            v8h k2 = *(const v8h*)(kp + 16);
            v8h k3 = *(const v8h*)(kp + 24);
            *(v8h*)&sK[krow][dc]      = k0;
            *(v8h*)&sK[krow][dc + 8]  = k1;
            *(v8h*)&sK[krow][dc + 16] = k2;
            *(v8h*)&sK[krow][dc + 24] = k3;
            const _Float16* vp = Vb + ((size_t)(j0 + krow)) * DM + dc;
            v8h v0 = *(const v8h*)vp;
            v8h v1 = *(const v8h*)(vp + 8);
            v8h v2 = *(const v8h*)(vp + 16);
            v8h v3 = *(const v8h*)(vp + 24);
#pragma unroll
            for (int i = 0; i < 8; ++i) {
                sVt[dc + i][krow]      = v0[i];
                sVt[dc + 8 + i][krow]  = v1[i];
                sVt[dc + 16 + i][krow] = v2[i];
                sVt[dc + 24 + i][krow] = v3[i];
            }
            if (j0 + BN < kend) {            // global_prefetch_b8 for next tile
                __builtin_prefetch(kp + (size_t)BN * DM, 0, 1);
                __builtin_prefetch(vp + (size_t)BN * DM, 0, 1);
            }
        }
        __syncthreads();

        if (j0 <= i0w + 15) {  // wave-uniform causal skip (EXEC stays all-1s)
            // --- S = Q @ K^T (four 16x16 column subtiles) ---
            v8f sacc[4];
#pragma unroll
            for (int s = 0; s < 4; ++s) sacc[s] = vzero;
#pragma unroll
            for (int ks = 0; ks < 4; ++ks) {
#pragma unroll
                for (int s = 0; s < 4; ++s) {
                    v16h kb = frag_ld(&sK[s * 16][0], DM + PADH, ks * 32, lane);
                    sacc[s] = wmma_f16(qf[ks], kb, sacc[s]);
                }
            }

            // --- online softmax (exp2 domain), causal mask, P -> LDS f16 ---
#pragma unroll
            for (int v = 0; v < 8; ++v) {
                int i = i0w + hi * 8 + v;
                float a[4];
                float tmax = -__builtin_inff();
#pragma unroll
                for (int s = 0; s < 4; ++s) {
                    int c = j0 + s * 16 + r;
                    a[s] = (c <= i) ? sacc[s][v] * sscale : -__builtin_inff();
                    tmax = fmaxf(tmax, a[s]);
                }
                float mt = rmax16(tmax);
                float mn = fmaxf(mrow[v], mt);
                float alpha = __builtin_exp2f(mrow[v] - mn);
                mrow[v] = mn;
                float p[4];
                float psum = 0.0f;
#pragma unroll
                for (int s = 0; s < 4; ++s) {
                    p[s] = __builtin_exp2f(a[s] - mn);
                    psum += p[s];
                }
                lrow[v] = lrow[v] * alpha + rsum16(psum);
#pragma unroll
                for (int s = 0; s < 8; ++s) o[s][v] *= alpha;
                int rl = hi * 8 + v;
#pragma unroll
                for (int s = 0; s < 4; ++s)
                    sP[wv][rl][s * 16 + r] = (_Float16)p[s];
            }

            // --- O += P @ V (K-dim 64 -> two slices) ---
            v16h pf0 = frag_ld(&sP[wv][0][0], BN + PADH, 0,  lane);
            v16h pf1 = frag_ld(&sP[wv][0][0], BN + PADH, 32, lane);
#pragma unroll
            for (int s = 0; s < 8; ++s) {
                v16h vb0 = frag_ld(&sVt[s * 16][0], BN + PADH, 0,  lane);
                v16h vb1 = frag_ld(&sVt[s * 16][0], BN + PADH, 32, lane);
                o[s] = wmma_f16(pf0, vb0, o[s]);
                o[s] = wmma_f16(pf1, vb1, o[s]);
            }
        }
        __syncthreads();
    }

    // --- epilogue: normalize and store fp32 ---
    float inv[8];
#pragma unroll
    for (int v = 0; v < 8; ++v) inv[v] = 1.0f / lrow[v];
    float* ob = out + ((size_t)b * SEQ + i0w) * DM;
#pragma unroll
    for (int s = 0; s < 8; ++s) {
        int c = s * 16 + r;
#pragma unroll
        for (int v = 0; v < 8; ++v) {
            ob[(size_t)(hi * 8 + v) * DM + c] = o[s][v] * inv[v];
        }
    }
}

// ---------------------------------------------------------------------------
extern "C" void kernel_launch(void* const* d_in, const int* in_sizes, int n_in,
                              void* d_out, int out_size, void* d_ws, size_t ws_size,
                              hipStream_t stream)
{
    (void)in_sizes; (void)n_in; (void)out_size; (void)ws_size;
    const float* x  = (const float*)d_in[0];
    const float* Wq = (const float*)d_in[1];
    const float* Wk = (const float*)d_in[2];
    const float* Wv = (const float*)d_in[3];
    float* out = (float*)d_out;

    size_t n = (size_t)BATCH * SEQ * DM;     // 8.4M elems -> 16MB f16 each
    _Float16* Qh = (_Float16*)d_ws;
    _Float16* Kh = Qh + n;
    _Float16* Vh = Kh + n;

    qkv_proj_kernel<<<dim3((BATCH * SEQ) / 128), 256, 0, stream>>>(
        x, Wq, Wk, Wv, Qh, Kh, Vh);
    attn_kernel<<<dim3(SEQ / BM, BATCH), 256, 0, stream>>>(
        Qh, Kh, Vh, out);
}
